// get_model_6897717478085
// MI455X (gfx1250) — matrix operand order repaired and checked by reading, not compiled, not run on hardware
//
#include <hip/hip_runtime.h>
#include <hip/hip_bf16.h>

typedef _Float16 f16;
typedef __attribute__((ext_vector_type(16))) _Float16 v16h;
typedef __attribute__((ext_vector_type(8)))  _Float16 v8h;
typedef __attribute__((ext_vector_type(8)))  float    v8f;

#define EPS_BN 1e-5f

// ---------------------------------------------------------------------------
// Weight prep: f32 (dout x din) -> f16 padded [doutp][dinp]; fold BN into
// per-channel scale/bias:  y = s*(x@W.T) + (s*b + beta),  s = gamma/sqrt(1+eps)
// ---------------------------------------------------------------------------
__global__ void prep_weights_kernel(const float* __restrict__ W, const float* __restrict__ b,
                                    const float* __restrict__ g, const float* __restrict__ be,
                                    f16* __restrict__ wq, float* __restrict__ scale,
                                    float* __restrict__ bias,
                                    int din, int dout, int dinp, int doutp) {
  int i = blockIdx.x * blockDim.x + threadIdx.x;
  int tot = dinp * doutp;
  if (i < tot) {
    int o = i / dinp, d = i % dinp;
    wq[i] = (o < dout && d < din) ? (f16)W[o * din + d] : (f16)0.0f;
  }
  if (i < doutp) {
    if (i < dout) {
      float s = g[i] * rsqrtf(1.0f + EPS_BN);
      scale[i] = s;
      bias[i]  = be[i] + b[i] * s;
    } else {
      scale[i] = 0.0f; bias[i] = 0.0f;
    }
  }
}

// xyz (B,6,N) -> pts (B,N,3), feats (B,N,3)
__global__ void split_xyz_kernel(const float* __restrict__ xyz, float* __restrict__ pts,
                                 float* __restrict__ feats, int B, int N) {
  int i = blockIdx.x * blockDim.x + threadIdx.x;
  if (i >= B * N) return;
  int b = i / N, n = i % N;
  for (int d = 0; d < 3; ++d) {
    pts[i * 3 + d]   = xyz[((size_t)b * 6 + d) * N + n];
    feats[i * 3 + d] = xyz[((size_t)b * 6 + 3 + d) * N + n];
  }
}

// ---------------------------------------------------------------------------
// Farthest point sampling: one block per batch, dist[] resident in LDS.
// Matches the jax scan (emit current farthest, update dist, argmax first-max).
// ---------------------------------------------------------------------------
__global__ void fps_kernel(const float* __restrict__ xyz, int* __restrict__ idx,
                           int N, int npoint) {
  __shared__ float dist[8192];
  __shared__ float rv[512];
  __shared__ int   ri[512];
  __shared__ int   sFar;
  int b = blockIdx.x, t = threadIdx.x, bd = blockDim.x;
  const float* x = xyz + (size_t)b * N * 3;
  for (int i = t; i < N; i += bd) dist[i] = 1e10f;
  if (t == 0) sFar = 0;
  __syncthreads();
  for (int it = 0; it < npoint; ++it) {
    int far = sFar;
    if (t == 0) idx[b * npoint + it] = far;
    float cx = x[far * 3], cy = x[far * 3 + 1], cz = x[far * 3 + 2];
    float bv = -1.0f; int bi = 0;
    for (int i = t; i < N; i += bd) {
      float dx = x[i * 3] - cx, dy = x[i * 3 + 1] - cy, dz = x[i * 3 + 2] - cz;
      float d2 = dx * dx + dy * dy + dz * dz;
      float nd = fminf(dist[i], d2);
      dist[i] = nd;
      if (nd > bv) { bv = nd; bi = i; }
    }
    rv[t] = bv; ri[t] = bi;
    __syncthreads();
    for (int s = bd >> 1; s > 0; s >>= 1) {
      if (t < s) {
        if (rv[t + s] > rv[t] || (rv[t + s] == rv[t] && ri[t + s] < ri[t])) {
          rv[t] = rv[t + s]; ri[t] = ri[t + s];
        }
      }
      __syncthreads();
    }
    if (t == 0) sFar = ri[0];
    __syncthreads();
  }
}

__global__ void gather_xyz_kernel(const float* __restrict__ xyz, const int* __restrict__ idx,
                                  float* __restrict__ nxyz, int N, int np, int Bnp) {
  int i = blockIdx.x * blockDim.x + threadIdx.x;
  if (i >= Bnp) return;
  int b = i / np;
  int g = idx[i];
  for (int d = 0; d < 3; ++d) nxyz[i * 3 + d] = xyz[((size_t)b * N + g) * 3 + d];
}

// Ball query: ascending-index scan == sorted-take-first-K of the reference;
// pad shortfall with first hit (centroid itself always qualifies, d2 == 0).
__global__ void ball_query_kernel(const float* __restrict__ xyz, const float* __restrict__ nxyz,
                                  int* __restrict__ gidx, int N, int np, int K, float r2, int Bnp) {
  int i = blockIdx.x * blockDim.x + threadIdx.x;
  if (i >= Bnp) return;
  int b = i / np;
  float cx = nxyz[i * 3], cy = nxyz[i * 3 + 1], cz = nxyz[i * 3 + 2];
  const float* x = xyz + (size_t)b * N * 3;
  int* out = gidx + (size_t)i * K;
  int cnt = 0;
  for (int j = 0; j < N && cnt < K; ++j) {
    float dx = x[j * 3] - cx, dy = x[j * 3 + 1] - cy, dz = x[j * 3 + 2] - cz;
    if (dx * dx + dy * dy + dz * dz <= r2) out[cnt++] = j;
  }
  int f = (cnt > 0) ? out[0] : 0;
  for (; cnt < K; ++cnt) out[cnt] = f;
}

// ---------------------------------------------------------------------------
// WMMA conv-BN-ReLU tile routine. 16x16x32 f16->f32 fragments in contiguous
// k order (A and B match):
//   A lane(c=lane/16, m=lane%16): 16 contiguous halves K = kt*32 + 16c + [0..15], row m
//   B lane(c, n=lane%16): 16 contiguous halves K = kt*32 + 16c + [0..15], col n
//   D lane(c, n): rows M = 8c + r, col n
// Works for both LDS-resident (ds_load_b128) and global activations.
// ---------------------------------------------------------------------------
__device__ __forceinline__ void wmma_layer(const f16* in, f16* out, const f16* __restrict__ w,
                                           const float* __restrict__ scale,
                                           const float* __restrict__ bias,
                                           int K, int kdimp, int doutp,
                                           int pitchIn, int pitchOut, int relu,
                                           int tid, int nthr) {
  int wave = tid >> 5, lane = tid & 31;
  int hi = lane >> 4, ln = lane & 15;
  int mtiles = K >> 4, ntiles = doutp >> 4, ktiles = kdimp >> 5;
  int nwaves = nthr >> 5;
  for (int tidx = wave; tidx < mtiles * ntiles; tidx += nwaves) {
    int i = tidx % mtiles, j = tidx / mtiles;
    v8f acc = {};
    const f16* arow = in + (i * 16 + ln) * pitchIn + hi * 16;
    const f16* brow = w + (size_t)(j * 16 + ln) * kdimp + hi * 16;
    for (int kt = 0; kt < ktiles; ++kt) {
      v8h a0 = *(const v8h*)(arow + kt * 32);
      v8h a1 = *(const v8h*)(arow + kt * 32 + 8);
      v16h av = __builtin_shufflevector(a0, a1, 0,1,2,3,4,5,6,7,8,9,10,11,12,13,14,15);
      v16h bv = *(const v16h*)(brow + kt * 32);
      acc = __builtin_amdgcn_wmma_f32_16x16x32_f16(false, av, false, bv,
                                                   (short)0, acc, false, false);
    }
    int o = j * 16 + ln;
    float sc = scale[o], bi = bias[o];
    f16* ocol = out + o;
#pragma unroll
    for (int r = 0; r < 8; ++r) {
      float v = acc[r] * sc + bi;
      if (relu) v = v > 0.0f ? v : 0.0f;
      ocol[(i * 16 + hi * 8 + r) * pitchOut] = (f16)v;
    }
  }
}

// ---------------------------------------------------------------------------
// One workgroup per centroid: gather grouped features into LDS, run the
// 3-layer MLP as LDS-resident WMMA GEMMs, max-pool over K samples.
// ---------------------------------------------------------------------------
__global__ void sa_branch_kernel(const float* __restrict__ pts, const float* __restrict__ feats,
                                 const float* __restrict__ nxyz, const int* __restrict__ gidx,
                                 const f16* __restrict__ w1, const float* __restrict__ s1, const float* __restrict__ b1,
                                 const f16* __restrict__ w2, const float* __restrict__ s2, const float* __restrict__ b2,
                                 const f16* __restrict__ w3, const float* __restrict__ s3, const float* __restrict__ b3,
                                 float* __restrict__ outF,
                                 int N, int cin, int np, int K,
                                 int d0p, int d1p, int d2p, int d3, int d3p,
                                 int Ctot, int coff, int pitch) {
  extern __shared__ __align__(32) f16 smem[];
  f16* bufA = smem;
  f16* bufB = smem + (size_t)K * pitch;
  int t = threadIdx.x, bd = blockDim.x;
  int bs = blockIdx.x;            // b*np + s
  int b = bs / np;
  const int* gi = gidx + (size_t)bs * K;
  float cx = nxyz[bs * 3], cy = nxyz[bs * 3 + 1], cz = nxyz[bs * 3 + 2];
  int d0 = cin + 3;

  __builtin_prefetch(w1, 0, 0);
  __builtin_prefetch(w2, 0, 0);
  __builtin_prefetch(w3, 0, 0);

  // gather: g = concat(feats[gidx], xyz[gidx] - new_xyz), zero-pad to pitch
  for (int i = t; i < K * pitch; i += bd) {
    int k = i / pitch, c = i % pitch;
    float v = 0.0f;
    if (c < d0) {
      int g = gi[k];
      if (c < cin) {
        v = feats[((size_t)b * N + g) * cin + c];
      } else {
        int d = c - cin;
        float pv = pts[((size_t)b * N + g) * 3 + d];
        v = pv - (d == 0 ? cx : (d == 1 ? cy : cz));
      }
    }
    bufA[i] = (f16)v;
  }
  __syncthreads();
  wmma_layer(bufA, bufB, w1, s1, b1, K, d0p, d1p, pitch, pitch, 1, t, bd);
  __syncthreads();
  wmma_layer(bufB, bufA, w2, s2, b2, K, d1p, d2p, pitch, pitch, 1, t, bd);
  __syncthreads();
  wmma_layer(bufA, bufB, w3, s3, b3, K, d2p, d3p, pitch, pitch, 1, t, bd);
  __syncthreads();
  // max-pool over the K samples
  for (int o = t; o < d3; o += bd) {
    float m = -1e30f;
    for (int k = 0; k < K; ++k) m = fmaxf(m, (float)bufB[k * pitch + o]);
    outF[(size_t)bs * Ctot + coff + o] = m;
  }
}

// MLP head layer: one block per 16-row M tile, waves split the N tiles.
__global__ void mlp_gemm_kernel(const f16* __restrict__ in, f16* __restrict__ out,
                                const f16* __restrict__ w, const float* __restrict__ scale,
                                const float* __restrict__ bias,
                                int kdimp, int doutp, int relu) {
  int i = blockIdx.x;
  wmma_layer(in + (size_t)i * 16 * kdimp, out + (size_t)i * 16 * doutp,
             w, scale, bias, 16, kdimp, doutp, kdimp, doutp, relu,
             threadIdx.x, blockDim.x);
}

__global__ void cvt_pad_kernel(const float* __restrict__ in, f16* __restrict__ out,
                               int M, int d, int dp) {
  int i = blockIdx.x * blockDim.x + threadIdx.x;
  if (i >= M * dp) return;
  int r = i / dp, c = i % dp;
  out[i] = (c < d) ? (f16)in[(size_t)r * d + c] : (f16)0.0f;
}

// softmax over s, then out[b,c,d] = sum_s w[b,s,c]*x[b,s,d]; one block per (b,c)
__global__ void softmax_head_kernel(const f16* __restrict__ h, const float* __restrict__ x,
                                    float* __restrict__ out, int S, int C, int hp) {
  __shared__ float red[256];
  int blk = blockIdx.x;
  int b = blk / C, c = blk % C;
  int t = threadIdx.x, bd = blockDim.x;
  float mx = -1e30f;
  for (int s = t; s < S; s += bd) mx = fmaxf(mx, (float)h[((size_t)b * S + s) * hp + c]);
  red[t] = mx; __syncthreads();
  for (int st = bd >> 1; st > 0; st >>= 1) {
    if (t < st) red[t] = fmaxf(red[t], red[t + st]);
    __syncthreads();
  }
  mx = red[0]; __syncthreads();
  float se = 0.0f, a0 = 0.0f, a1 = 0.0f, a2 = 0.0f;
  for (int s = t; s < S; s += bd) {
    float e = expf((float)h[((size_t)b * S + s) * hp + c] - mx);
    se += e;
    a0 += e * x[((size_t)b * S + s) * 3 + 0];
    a1 += e * x[((size_t)b * S + s) * 3 + 1];
    a2 += e * x[((size_t)b * S + s) * 3 + 2];
  }
  float vals[4] = {se, a0, a1, a2};
  float res[4];
  for (int q = 0; q < 4; ++q) {
    red[t] = vals[q]; __syncthreads();
    for (int st = bd >> 1; st > 0; st >>= 1) {
      if (t < st) red[t] += red[t + st];
      __syncthreads();
    }
    res[q] = red[0]; __syncthreads();
  }
  if (t == 0) {
    float inv = 1.0f / res[0];
    size_t o = ((size_t)b * C + c) * 3;
    out[o + 0] = res[1] * inv;
    out[o + 1] = res[2] * inv;
    out[o + 2] = res[3] * inv;
  }
}

// ---------------------------------------------------------------------------
// Host orchestration
// ---------------------------------------------------------------------------
extern "C" void kernel_launch(void* const* d_in, const int* in_sizes, int n_in,
                              void* d_out, int out_size, void* d_ws, size_t ws_size,
                              hipStream_t stream) {
  (void)in_sizes; (void)n_in; (void)out_size; (void)ws_size;
  const int B = 4, N0 = 8192;

  char* wsp = (char*)d_ws;
  size_t wo = 0;
  auto alloc = [&](size_t bytes) -> void* {
    void* p = wsp + wo;
    wo += (bytes + 255) & ~(size_t)255;
    return p;
  };
  auto pad32 = [](int x) { return (x + 31) & ~31; };

  int ii = 0;
  const float* xyz = (const float*)d_in[ii++];

  struct LP { const float *W, *b, *g, *be; int din, dout; };
  LP lp[29];
  int nl = 0;
  const int cins[4] = {3, 64, 128, 256};
  const int mdims[4][3] = {{16,16,32},{32,32,64},{64,64,128},{128,128,256}};
  for (int st = 0; st < 4; ++st)
    for (int br = 0; br < 2; ++br) {
      int d = cins[st] + 3;
      for (int l = 0; l < 3; ++l) {
        LP p;
        p.W  = (const float*)d_in[ii++];
        p.b  = (const float*)d_in[ii++];
        p.g  = (const float*)d_in[ii++];
        p.be = (const float*)d_in[ii++];
        p.din = d; p.dout = mdims[st][l];
        d = p.dout;
        lp[nl++] = p;
      }
    }
  const int mlpd[6] = {512, 384, 256, 256, 128, 100};
  for (int l = 0; l < 5; ++l) {
    LP p;
    p.W  = (const float*)d_in[ii++];
    p.b  = (const float*)d_in[ii++];
    p.g  = (const float*)d_in[ii++];
    p.be = (const float*)d_in[ii++];
    p.din = mlpd[l]; p.dout = mlpd[l + 1];
    lp[nl++] = p;
  }
  // d_in[ii] = num_class (unused)

  // weight prep (f16 + folded BN scale/bias)
  f16*   wq[29];
  float* wsc[29];
  float* wbi[29];
  for (int l = 0; l < 29; ++l) {
    int dip = pad32(lp[l].din), dop = pad32(lp[l].dout);
    wq[l]  = (f16*)alloc((size_t)dip * dop * sizeof(f16));
    wsc[l] = (float*)alloc((size_t)dop * 4);
    wbi[l] = (float*)alloc((size_t)dop * 4);
    int tot = dip * dop;
    prep_weights_kernel<<<(tot + 255) / 256, 256, 0, stream>>>(
        lp[l].W, lp[l].b, lp[l].g, lp[l].be, wq[l], wsc[l], wbi[l],
        lp[l].din, lp[l].dout, dip, dop);
  }

  float* pts0   = (float*)alloc((size_t)B * N0 * 3 * 4);
  float* feats0 = (float*)alloc((size_t)B * N0 * 3 * 4);
  split_xyz_kernel<<<(B * N0 + 255) / 256, 256, 0, stream>>>(xyz, pts0, feats0, B, N0);

  const int   NP[4]     = {1024, 768, 512, 512};
  const float RAD[4][2] = {{0.1f,0.2f},{0.2f,0.4f},{0.4f,0.6f},{0.6f,0.8f}};
  const int   KS[4][2]  = {{16,32},{32,64},{32,64},{64,128}};

  float* curXyz = pts0;
  float* curF   = feats0;
  int N = N0, curC = 3, lidx = 0;
  float* nxyzFinal = nullptr;

  for (int st = 0; st < 4; ++st) {
    int np = NP[st];
    int Bnp = B * np;
    int* fpsIdx = (int*)alloc((size_t)Bnp * 4);
    fps_kernel<<<B, 512, 0, stream>>>(curXyz, fpsIdx, N, np);
    float* nxyz = (float*)alloc((size_t)Bnp * 3 * 4);
    gather_xyz_kernel<<<(Bnp + 255) / 256, 256, 0, stream>>>(curXyz, fpsIdx, nxyz, N, np, Bnp);

    int Ctot = 2 * mdims[st][2];
    float* fout = (float*)alloc((size_t)Bnp * Ctot * 4);

    int d0 = cins[st] + 3;
    int pitch = pad32(d0);
    for (int l = 0; l < 3; ++l) { int p = pad32(mdims[st][l]); if (p > pitch) pitch = p; }

    int coff = 0;
    for (int br = 0; br < 2; ++br) {
      int K = KS[st][br];
      float r = RAD[st][br];
      int* gi = (int*)alloc((size_t)Bnp * K * 4);
      ball_query_kernel<<<(Bnp + 127) / 128, 128, 0, stream>>>(
          curXyz, nxyz, gi, N, np, K, r * r, Bnp);

      int d0p = pad32(lp[lidx].din);
      int d1p = pad32(lp[lidx].dout);
      int d2p = pad32(lp[lidx + 1].dout);
      int d3  = lp[lidx + 2].dout;
      int d3p = pad32(d3);
      size_t sh = (size_t)2 * K * pitch * sizeof(f16);
      sa_branch_kernel<<<Bnp, 128, sh, stream>>>(
          curXyz, curF, nxyz, gi,
          wq[lidx],     wsc[lidx],     wbi[lidx],
          wq[lidx + 1], wsc[lidx + 1], wbi[lidx + 1],
          wq[lidx + 2], wsc[lidx + 2], wbi[lidx + 2],
          fout, N, curC, np, K, d0p, d1p, d2p, d3, d3p, Ctot, coff, pitch);
      coff += d3;
      lidx += 3;
    }
    curXyz = nxyz; curF = fout; N = np; curC = Ctot;
    nxyzFinal = nxyz;
  }

  // MLP head over the 512 surviving points (M = 2048 rows)
  int M = B * 512;
  f16* actA = (f16*)alloc((size_t)M * 512 * sizeof(f16));
  f16* actB = (f16*)alloc((size_t)M * 512 * sizeof(f16));
  cvt_pad_kernel<<<(M * 512 + 255) / 256, 256, 0, stream>>>(curF, actA, M, 512, 512);
  f16* cur = actA;
  f16* nxt = actB;
  for (int l = 0; l < 5; ++l) {
    int dip = pad32(mlpd[l]), dop = pad32(mlpd[l + 1]);
    mlp_gemm_kernel<<<M / 16, 128, 0, stream>>>(
        cur, nxt, wq[24 + l], wsc[24 + l], wbi[24 + l], dip, dop, (l < 4) ? 1 : 0);
    f16* tmp = cur; cur = nxt; nxt = tmp;
  }

  softmax_head_kernel<<<B * 100, 256, 0, stream>>>(
      cur, nxyzFinal, (float*)d_out, 512, 100, pad32(100));
}
